// SparseBasicBlock_45981919871118
// MI455X (gfx1250) — matrix operand order repaired and checked by reading, not compiled, not run on hardware
//
#include <hip/hip_runtime.h>
#include <hip/hip_bf16.h>

#define C     64
#define KOFF  27
#define BN_EPS 1e-3f
#define STATB 240   // bn-stats grid

typedef __attribute__((ext_vector_type(16))) _Float16 v16h;
typedef __attribute__((ext_vector_type(8)))  _Float16 v8h;
typedef __attribute__((ext_vector_type(8)))  float    v8f;

union V16 { v16h v; v8h h[2]; };

static __device__ __forceinline__ void wait_async0() {
#if __has_builtin(__builtin_amdgcn_s_wait_asynccnt)
  __builtin_amdgcn_s_wait_asynccnt(0);
#else
  asm volatile("s_wait_asynccnt 0x0" ::: "memory");
#endif
}

// Async copy of 32 bytes global -> LDS (two b128s; INST_OFFSET applies to both
// the LDS and the memory address per the CDNA5 async-load pseudocode).
static __device__ __forceinline__ void async_copy32(uint32_t lds_off, uint64_t gaddr) {
  asm volatile("global_load_async_to_lds_b128 %0, %1, off\n\t"
               "global_load_async_to_lds_b128 %0, %1, off offset:16"
               :: "v"(lds_off), "v"(gaddr) : "memory");
}

// ---------------------------------------------------------------- prep kernels
__global__ void feat_to_h_kernel(const float* __restrict__ F,
                                 _Float16* __restrict__ Xh, int N) {
  int i = blockIdx.x * blockDim.x + threadIdx.x;
  int total = (N + 1) * C;                 // row N is the zero sentinel row
  if (i >= total) return;
  Xh[i] = (i < N * C) ? (_Float16)F[i] : (_Float16)0.0f;
}

__global__ void w_to_h_kernel(const float* __restrict__ w1,
                              const float* __restrict__ w2,
                              _Float16* __restrict__ W1h,
                              _Float16* __restrict__ W2h) {
  int i = blockIdx.x * blockDim.x + threadIdx.x;
  if (i >= KOFF * C * C) return;
  W1h[i] = (_Float16)w1[i];
  W2h[i] = (_Float16)w2[i];
}

// ---------------------------------------------------------------- conv (WMMA)
// Y[N,64] = sum_k gather(Xh, nbr[k]) @ Wh[k]   (Xh padded with zero row N)
// wave: 32 rows x 64 cout; 16 v_wmma_f32_16x16x32_f16 per offset.
// W[k] (8 KB f16) is double-buffered in LDS via async global->LDS copies.
__global__ __launch_bounds__(256)
void subm_conv_wmma_kernel(const _Float16* __restrict__ Xh,
                           const int* __restrict__ nbr,
                           const _Float16* __restrict__ Wh,
                           float* __restrict__ Y, int N) {
  __shared__ __align__(16) _Float16 sW[2][C * C];   // 16 KB double buffer

  const int t    = threadIdx.x;
  const int lane = t & 31;
  const int wave = t >> 5;
  const int mrow = lane & 15;          // M row / N col within a 16x16 tile
  const int hi   = lane >> 4;          // 0 or 1
  const int kb   = hi * 8;             // A-matrix K sub-offset (per ISA layout)
  const int base = (blockIdx.x * 8 + wave) * 32;

  // LDS byte offsets of this thread's 32-byte copy chunk in each buffer
  const uint32_t lds0 = (uint32_t)(uintptr_t)(&sW[0][t * 16]);
  const uint32_t lds1 = (uint32_t)(uintptr_t)(&sW[1][t * 16]);

  // stage W[0] into buffer 0
  async_copy32(lds0, (uint64_t)(uintptr_t)(Wh + t * 16));
  wait_async0();
  __syncthreads();

  v8f acc[8] = {};                     // [mt*4 + nt], each 16x16 f32 tile

  const int r0 = base + mrow;          // M-tile 0 row for this lane
  const int r1 = base + 16 + mrow;     // M-tile 1 row
  const int Nm1 = N - 1;

  for (int k = 0; k < KOFF; ++k) {
    const int cur = k & 1;

    // overlap: async-prefetch W[k+1] into the other buffer
    if (k + 1 < KOFF) {
      async_copy32(cur ? lds0 : lds1,
                   (uint64_t)(uintptr_t)(Wh + (size_t)(k + 1) * C * C + t * 16));
    }

    // neighbor indices: clamped unconditional load + select (no EXEC branches)
    const int* nk = nbr + (size_t)k * N;
    int idx0 = nk[r0 < Nm1 ? r0 : Nm1];
    int idx1 = nk[r1 < Nm1 ? r1 : Nm1];
    idx0 = (r0 < N) ? idx0 : N;        // sentinel -> zero row
    idx1 = (r1 < N) ? idx1 : N;
    const _Float16* x0 = Xh + (size_t)idx0 * C;
    const _Float16* x1 = Xh + (size_t)idx1 * C;

    // A tiles: lane holds K runs [kb,kb+8) and [kb+16,kb+24) of its row,
    // for K-chunk 0 (cin 0..31) and chunk 1 (cin 32..63).
    V16 a00, a01, a10, a11;
    a00.h[0] = *(const v8h*)(x0 + kb);
    a00.h[1] = *(const v8h*)(x0 + kb + 16);
    a01.h[0] = *(const v8h*)(x0 + kb + 32);
    a01.h[1] = *(const v8h*)(x0 + kb + 48);
    a10.h[0] = *(const v8h*)(x1 + kb);
    a10.h[1] = *(const v8h*)(x1 + kb + 16);
    a11.h[0] = *(const v8h*)(x1 + kb + 32);
    a11.h[1] = *(const v8h*)(x1 + kb + 48);

    // B tiles from LDS: lane holds col cout = nt*16 + mrow, K runs [kb,kb+8) and [kb+16,kb+24) along cin
    const _Float16* b0 = &sW[cur][mrow];
    const _Float16* b1 = &sW[cur][32 * C + mrow];

#pragma unroll
    for (int nt = 0; nt < 4; ++nt) {
      V16 B0, B1;
#pragma unroll
      for (int s = 0; s < 16; ++s) {
        const int kk = (s < 8) ? (kb + s) : (kb + 8 + s);
        B0.v[s] = b0[kk * C + nt * 16];
        B1.v[s] = b1[kk * C + nt * 16];
      }
      acc[nt]     = __builtin_amdgcn_wmma_f32_16x16x32_f16(false, a00.v, false, B0.v,
                        (short)0, acc[nt],     false, false);
      acc[nt]     = __builtin_amdgcn_wmma_f32_16x16x32_f16(false, a01.v, false, B1.v,
                        (short)0, acc[nt],     false, false);
      acc[nt + 4] = __builtin_amdgcn_wmma_f32_16x16x32_f16(false, a10.v, false, B0.v,
                        (short)0, acc[nt + 4], false, false);
      acc[nt + 4] = __builtin_amdgcn_wmma_f32_16x16x32_f16(false, a11.v, false, B1.v,
                        (short)0, acc[nt + 4], false, false);
    }

    // this wave's async copy of W[k+1] must have landed before next iteration
    wait_async0();
    __syncthreads();
  }

  // D layout: lane -> col = mrow; VGPR v -> row = v + 8*hi (+16 for M-tile 1)
#pragma unroll
  for (int mt = 0; mt < 2; ++mt) {
    const int rb = base + mt * 16 + hi * 8;
#pragma unroll
    for (int nt = 0; nt < 4; ++nt) {
#pragma unroll
      for (int v = 0; v < 8; ++v) {
        const int row = rb + v;
        if (row < N) Y[(size_t)row * C + nt * 16 + mrow] = acc[mt * 4 + nt][v];
      }
    }
  }
}

// ---------------------------------------------------------------- batchnorm
__global__ void bn_stats_kernel(const float* __restrict__ Y,
                                float* __restrict__ part, int N) {
  __shared__ float ss[256], sq[256];
  const int c   = threadIdx.x & 63;
  const int sub = threadIdx.x >> 6;     // 0..3
  float s = 0.f, q = 0.f;
  for (int r = blockIdx.x * 4 + sub; r < N; r += gridDim.x * 4) {
    const float v = Y[(size_t)r * C + c];
    s += v; q += v * v;
  }
  ss[threadIdx.x] = s; sq[threadIdx.x] = q;
  __syncthreads();
  if (threadIdx.x < 64) {
    s = ss[c] + ss[c + 64] + ss[c + 128] + ss[c + 192];
    q = sq[c] + sq[c + 64] + sq[c + 128] + sq[c + 192];
    part[blockIdx.x * 128 + c]      = s;
    part[blockIdx.x * 128 + 64 + c] = q;
  }
}

__global__ void bn_finalize_kernel(const float* __restrict__ part, int nblocks,
                                   const float* __restrict__ gamma,
                                   const float* __restrict__ beta,
                                   float* __restrict__ scsh, int N) {
  const int c = threadIdx.x;
  if (c >= C) return;
  float s = 0.f, q = 0.f;
  for (int b = 0; b < nblocks; ++b) {
    s += part[b * 128 + c];
    q += part[b * 128 + 64 + c];
  }
  const float inv = 1.0f / (float)N;
  const float mu  = s * inv;
  const float var = q * inv - mu * mu;
  const float sc  = gamma[c] * rsqrtf(var + BN_EPS);
  scsh[c]     = sc;
  scsh[C + c] = beta[c] - mu * sc;
}

__global__ void bn_relu_to_h_kernel(const float* __restrict__ Y,
                                    const float* __restrict__ scsh,
                                    _Float16* __restrict__ Zh, int N) {
  int i = blockIdx.x * blockDim.x + threadIdx.x;
  int total = (N + 1) * C;
  if (i >= total) return;
  if (i >= N * C) { Zh[i] = (_Float16)0.0f; return; }
  const int c = i & 63;
  const float v = fmaf(Y[i], scsh[c], scsh[C + c]);
  Zh[i] = (_Float16)fmaxf(v, 0.0f);
}

__global__ void final_kernel(const float* __restrict__ Y,
                             const float* __restrict__ scsh,
                             const float* __restrict__ F,
                             float* __restrict__ out, int N) {
  int i = blockIdx.x * blockDim.x + threadIdx.x;
  if (i >= N * C) return;
  const int c = i & 63;
  const float v = fmaf(Y[i], scsh[c], scsh[C + c]) + F[i];
  out[i] = fmaxf(v, 0.0f);
}

// ---------------------------------------------------------------- launcher
extern "C" void kernel_launch(void* const* d_in, const int* in_sizes, int n_in,
                              void* d_out, int out_size, void* d_ws, size_t ws_size,
                              hipStream_t stream) {
  const float* features  = (const float*)d_in[0];
  const int*   neighbors = (const int*)  d_in[1];
  const float* w1 = (const float*)d_in[2];
  const float* g1 = (const float*)d_in[3];
  const float* b1 = (const float*)d_in[4];
  const float* w2 = (const float*)d_in[5];
  const float* g2 = (const float*)d_in[6];
  const float* b2 = (const float*)d_in[7];
  float* out = (float*)d_out;

  const int N = in_sizes[0] / C;

  char* ws = (char*)d_ws;
  size_t off = 0;
  auto alloc = [&](size_t bytes) -> void* {
    void* p = ws + off;
    off = (off + bytes + 255) & ~(size_t)255;
    return p;
  };
  _Float16* Xh   = (_Float16*)alloc((size_t)(N + 1) * C * sizeof(_Float16)); // also reused as Z1h
  _Float16* W1h  = (_Float16*)alloc((size_t)KOFF * C * C * sizeof(_Float16));
  _Float16* W2h  = (_Float16*)alloc((size_t)KOFF * C * C * sizeof(_Float16));
  float*    Y    = (float*)   alloc((size_t)N * C * sizeof(float));          // Y1 then Y2
  float*    part = (float*)   alloc((size_t)STATB * 128 * sizeof(float));
  float*    scsh = (float*)   alloc((size_t)2 * C * sizeof(float));
  (void)ws_size; (void)n_in; (void)out_size;

  const int totalPad = (N + 1) * C;
  const int convBlocks = (N + 255) / 256;   // 8 waves x 32 rows per block

  feat_to_h_kernel<<<(totalPad + 255) / 256, 256, 0, stream>>>(features, Xh, N);
  w_to_h_kernel<<<(KOFF * C * C + 255) / 256, 256, 0, stream>>>(w1, w2, W1h, W2h);

  // conv1 -> BN -> ReLU -> f16
  subm_conv_wmma_kernel<<<convBlocks, 256, 0, stream>>>(Xh, neighbors, W1h, Y, N);
  bn_stats_kernel<<<STATB, 256, 0, stream>>>(Y, part, N);
  bn_finalize_kernel<<<1, 64, 0, stream>>>(part, STATB, g1, b1, scsh, N);
  bn_relu_to_h_kernel<<<(totalPad + 255) / 256, 256, 0, stream>>>(Y, scsh, Xh, N);

  // conv2 -> BN -> +residual -> ReLU
  subm_conv_wmma_kernel<<<convBlocks, 256, 0, stream>>>(Xh, neighbors, W2h, Y, N);
  bn_stats_kernel<<<STATB, 256, 0, stream>>>(Y, part, N);
  bn_finalize_kernel<<<1, 64, 0, stream>>>(part, STATB, g2, b2, scsh, N);
  final_kernel<<<(N * C + 255) / 256, 256, 0, stream>>>(Y, scsh, features, out, N);
}
